// Spatial_Attention_18391049961784
// MI455X (gfx1250) — matrix operand order repaired and checked by reading, not compiled, not run on hardware
//
#include <hip/hip_runtime.h>

// ---------------------------------------------------------------------------
// CSWin windowed attention for MI455X (gfx1250, wave32, WMMA f16 16x16x32)
// 12288 blocks = 2048 windows x 6 heads; 128 threads = 4 waves; wave w owns
// query rows [16w, 16w+16).  Computes S^T = K Q^T, in-lane softmax, then
// Z^T = V^T P^T using shfl-built B operands.  HBM-bound (~400MB -> ~17us).
// ---------------------------------------------------------------------------

typedef __attribute__((ext_vector_type(16))) _Float16 v16h;
typedef __attribute__((ext_vector_type(8)))  float    v8f;
typedef __attribute__((ext_vector_type(4)))  unsigned int u32x4;
typedef __attribute__((ext_vector_type(4)))  float    f32x4;

#define NUM_HEADS 6
#define HD   32            // head dim
#define WIN  64            // tokens per 8x8 window
#define HH   128
#define WW   128
#define CC   192
#define LL   (HH * WW)
#define QSTRIDE 40         // 32 + 8 pad halves (80B rows, 16B aligned)
#define VSTRIDE 72         // 64 + 8 pad halves (144B rows, 16B aligned)

union V16H  { v16h v; u32x4 q[2]; _Float16 h[16]; };
union HPack { u32x4 u; _Float16 h[8]; };

__global__ __launch_bounds__(128) void cswin_attn_kernel(
    const float* __restrict__ qkv1,
    const float* __restrict__ qkv2,
    float* __restrict__ out)
{
    __shared__ _Float16 sQ[WIN * QSTRIDE];   // Q  (64 x 32 f16, row-major)
    __shared__ _Float16 sK[WIN * QSTRIDE];   // K  (64 x 32 f16, row-major)
    __shared__ _Float16 sV[HD  * VSTRIDE];   // V^T (32 x 64 f16, row-major)

    const int tid  = threadIdx.x;
    const int bid  = blockIdx.x;
    const int head = bid % NUM_HEADS;
    const int win  = bid / NUM_HEADS;
    const int b    = win >> 8;               // 16x16 = 256 windows / batch
    const int rem  = win & 255;
    const int wy   = rem >> 4;
    const int wx   = rem & 15;

    const size_t PLANE = (size_t)8 * LL * CC;     // one (B,L,C) plane
    const float* qbase = qkv1;                    // qkv1[0] = q1
    const float* vbase = qkv1 + 2 * PLANE;        // qkv1[2] = v1
    const float* kbase = qkv2 + 1 * PLANE;        // qkv2[1] = k2

    // ---------------- stage: global f32 -> LDS f16 (V transposed) ----------
    {
        const int tok = tid >> 1;                 // 0..63
        const int hf  = tid & 1;                  // half of the 32-ch row
        const int ty  = tok >> 3, tx = tok & 7;
        const size_t l = (size_t)(wy * 8 + ty) * WW + wx * 8 + tx;
        const size_t rowoff = ((size_t)b * LL + l) * CC + head * HD + hf * 16;

        const f32x4* gq = (const f32x4*)(qbase + rowoff);
        const f32x4* gk = (const f32x4*)(kbase + rowoff);
        const f32x4* gv = (const f32x4*)(vbase + rowoff);

        float qe[16], ke[16], ve[16];
        #pragma unroll
        for (int i = 0; i < 4; ++i) {
            ((f32x4*)qe)[i] = gq[i];
            ((f32x4*)ke)[i] = gk[i];
            ((f32x4*)ve)[i] = gv[i];
        }

        HPack pq0, pq1, pk0, pk1;
        #pragma unroll
        for (int i = 0; i < 8; ++i) {
            pq0.h[i] = (_Float16)qe[i];  pq1.h[i] = (_Float16)qe[8 + i];
            pk0.h[i] = (_Float16)ke[i];  pk1.h[i] = (_Float16)ke[8 + i];
        }
        u32x4* dq = (u32x4*)(sQ + tok * QSTRIDE + hf * 16);
        u32x4* dk = (u32x4*)(sK + tok * QSTRIDE + hf * 16);
        dq[0] = pq0.u; dq[1] = pq1.u;
        dk[0] = pk0.u; dk[1] = pk1.u;

        const int d0 = hf * 16;                   // V transposed: sV[d][tok]
        #pragma unroll
        for (int i = 0; i < 16; ++i)
            sV[(d0 + i) * VSTRIDE + tok] = (_Float16)ve[i];
    }
    __syncthreads();

    // ---------------- per-wave attention -----------------------------------
    const int wv    = tid >> 5;                   // wave id: query tile
    const int lane  = tid & 31;
    const int lhalf = lane >> 4;                  // half-wave (0/1)
    const int lm    = lane & 15;                  // N index (query col m)

    // B operand: Q^T.  lane -> q row m = 16wv+lm, d in [16*lhalf, +16)
    V16H bq;
    {
        const _Float16* qs = sQ + (wv * 16 + lm) * QSTRIDE + 16 * lhalf;
        bq.q[0] = *(const u32x4*)qs;
        bq.q[1] = *(const u32x4*)(qs + 8);
    }

    // S^T[jt] = K_tile[jt] (16x32) @ Q^T (32x16), f32 accum
    v8f st[4];
    #pragma unroll
    for (int jt = 0; jt < 4; ++jt) {
        V16H ak;
        const _Float16* kr = sK + (jt * 16 + lm) * QSTRIDE;
        ak.q[0] = *(const u32x4*)(kr + 16 * lhalf);
        ak.q[1] = *(const u32x4*)(kr + 16 * lhalf + 8);
        v8f cz = (v8f)0.0f;
        st[jt] = __builtin_amdgcn_wmma_f32_16x16x32_f16(
            false, ak.v, false, bq.v, (short)0, cz, false, false);
    }

    // softmax over j for fixed m (in-lane 32 values + half-wave partner)
    const float sc = 0.17677669529663687f;        // 1/sqrt(32)
    float pv[4][8];
    float mx = -3.0e38f;
    #pragma unroll
    for (int jt = 0; jt < 4; ++jt)
        #pragma unroll
        for (int r = 0; r < 8; ++r) {
            float x = st[jt][r] * sc;
            pv[jt][r] = x;
            mx = fmaxf(mx, x);
        }
    mx = fmaxf(mx, __shfl_xor(mx, 16, 32));
    float sum = 0.0f;
    #pragma unroll
    for (int jt = 0; jt < 4; ++jt)
        #pragma unroll
        for (int r = 0; r < 8; ++r) {
            float e = exp2f((pv[jt][r] - mx) * 1.4426950408889634f);
            pv[jt][r] = e;
            sum += e;
        }
    sum += __shfl_xor(sum, 16, 32);
    const float inv = 1.0f / sum;

    // Build P^T as WMMA B operands (K chunks of 32 j values) via shfl_xor(16)
    V16H bp[2];
    #pragma unroll
    for (int c = 0; c < 2; ++c) {
        #pragma unroll
        for (int h = 0; h < 8; ++h) {
            float oA = __shfl_xor(pv[2 * c][h],     16, 32);
            float oB = __shfl_xor(pv[2 * c + 1][h], 16, 32);
            bp[c].h[h]     = (_Float16)(lhalf ? oB : pv[2 * c][h]);
            bp[c].h[h + 8] = (_Float16)(lhalf ? pv[2 * c + 1][h] : oA);
        }
    }

    // Z^T[dt] = V^T (16x64) @ P^T (64x16), K split into two 32-chunks
    v8f zt[2] = { (v8f)0.0f, (v8f)0.0f };
    #pragma unroll
    for (int dt = 0; dt < 2; ++dt) {
        #pragma unroll
        for (int c = 0; c < 2; ++c) {
            V16H av;
            const _Float16* vr = sV + (dt * 16 + lm) * VSTRIDE + 32 * c;
            av.q[0] = *(const u32x4*)(vr + 16 * lhalf);
            av.q[1] = *(const u32x4*)(vr + 16 * lhalf + 8);
            zt[dt] = __builtin_amdgcn_wmma_f32_16x16x32_f16(
                false, av.v, false, bp[c].v, (short)0, zt[dt], false, false);
        }
    }

    // ---------------- write out: lane = token m, 8 consecutive channels ----
    const int m  = wv * 16 + lm;
    const int ty = m >> 3, tx = m & 7;
    const size_t lpos = (size_t)(wy * 8 + ty) * WW + wx * 8 + tx;
    #pragma unroll
    for (int dt = 0; dt < 2; ++dt) {
        float o[8];
        #pragma unroll
        for (int r = 0; r < 8; ++r) o[r] = zt[dt][r] * inv;
        float* op = out + ((size_t)b * LL + lpos) * CC
                        + head * HD + dt * 16 + 8 * lhalf;
        ((f32x4*)op)[0] = ((f32x4*)o)[0];
        ((f32x4*)op)[1] = ((f32x4*)o)[1];
    }
}

extern "C" void kernel_launch(void* const* d_in, const int* in_sizes, int n_in,
                              void* d_out, int out_size, void* d_ws, size_t ws_size,
                              hipStream_t stream) {
    const float* qkv1 = (const float*)d_in[0];
    const float* qkv2 = (const float*)d_in[1];
    // d_in[2], d_in[3] are scalar H, W (128) -> compile-time constants here
    float* out = (float*)d_out;
    (void)in_sizes; (void)n_in; (void)out_size; (void)d_ws; (void)ws_size;

    dim3 grid(2048 * NUM_HEADS);   // 12288 (window, head) problems
    dim3 block(128);               // 4 wave32 waves
    hipLaunchKernelGGL(cswin_attn_kernel, grid, block, 0, stream,
                       qkv1, qkv2, out);
}
